// HeteroGNNLayer_21251498180621
// MI455X (gfx1250) — matrix operand order repaired and checked by reading, not compiled, not run on hardware
//
#include <hip/hip_runtime.h>
#include <hip/hip_bf16.h>

// H=4 heads, C=64 ch/head, D=256 input dim, out = H*C = 256.
#define DDIM 256
#define LDS_STRIDE 272   // bf16 elements per LDS row (32B pad; 544B row pitch keeps 16B align)

typedef __attribute__((ext_vector_type(16))) __bf16 v16bf;
typedef __attribute__((ext_vector_type(8)))  float  v8f;

struct U4x2 { uint4 lo, hi; };
static __device__ __forceinline__ v16bf frag_from(uint4 lo, uint4 hi) {
    U4x2 t; t.lo = lo; t.hi = hi;
    return __builtin_bit_cast(v16bf, t);     // 32B -> 16 x bf16
}

static __device__ __forceinline__ unsigned short f2bf(float f) {
    unsigned u = __float_as_uint(f);
    u += 0x7FFFu + ((u >> 16) & 1u);          // round-to-nearest-even
    return (unsigned short)(u >> 16);
}
// Monotone order-preserving float->uint encoding for atomicMax on floats.
static __device__ __forceinline__ unsigned encf(float f) {
    unsigned u = __float_as_uint(f);
    return (u & 0x80000000u) ? ~u : (u | 0x80000000u);
}
static __device__ __forceinline__ float decf(unsigned u) {
    return (u & 0x80000000u) ? __uint_as_float(u & 0x7FFFFFFFu) : __uint_as_float(~u);
}

__global__ void k_zero(float* __restrict__ p, int n4) {
    int i = blockIdx.x * blockDim.x + threadIdx.x;
    if (i < n4) ((float4*)p)[i] = make_float4(0.f, 0.f, 0.f, 0.f);
}

__global__ void k_init_md(unsigned* __restrict__ m, float* __restrict__ den, int n) {
    int i = blockIdx.x * blockDim.x + threadIdx.x;
    if (i < n) { m[i] = 0x007FFFFFu; /* encf(-inf) */ den[i] = 0.f; }
}

// One-time f32 -> packed bf16 conversion (2 elems/thread).
__global__ void k_cvt_bf16(const float* __restrict__ X, unsigned* __restrict__ Xb2, int n2) {
    int i = blockIdx.x * blockDim.x + threadIdx.x;
    if (i >= n2) return;
    float2 v = ((const float2*)X)[i];
    Xb2[i] = (unsigned)f2bf(v.x) | ((unsigned)f2bf(v.y) << 16);
}

// Pre-pack W (256x256 f32, row-major [k][n]) into B-fragment order:
// group g = (kk*256 + n)*2 + half  ->  16 contiguous bf16: W[kk*32 + half*16 + i][n].
__global__ void k_pack_w(const float* __restrict__ W, unsigned short* __restrict__ Wp) {
    int g = blockIdx.x * blockDim.x + threadIdx.x;
    if (g >= 4096) return;
    int half = g & 1;
    int n    = (g >> 1) & 255;
    int kk   = g >> 9;
    int kb   = kk * 32 + half * 16;
    unsigned short* o = Wp + (size_t)g * 16;
    #pragma unroll
    for (int i = 0; i < 16; ++i) o[i] = f2bf(W[(size_t)(kb + i) * DDIM + n]);
}

// h = x @ W : (n_rows x 256) @ (256 x 256), bf16 in / f32 accumulate.
// Block: 512 threads = 16 waves; block owns 16-row M tile, wave w owns cols [16w,16w+16).
__global__ void __launch_bounds__(512) k_gemm(const unsigned short* __restrict__ Xb,
                                              const unsigned short* __restrict__ Wp,
                                              float* __restrict__ Hout, int n_rows) {
    __shared__ unsigned short sA[16 * LDS_STRIDE];
    const int tid = threadIdx.x;
    const int rowBase = blockIdx.x * 16;

    {   // stage 16x256 bf16 tile: one b128 per thread
        int row = tid >> 5;
        int c8  = (tid & 31) * 8;                       // ushort offset in row
        int gr  = rowBase + row; if (gr > n_rows - 1) gr = n_rows - 1;
        uint4 v = *(const uint4*)(Xb + (size_t)gr * DDIM + c8);
        *(uint4*)(&sA[row * LDS_STRIDE + c8]) = v;
    }
    __syncthreads();

    const int lane = tid & 31;
    const int wave = tid >> 5;
    const int colBase = wave * 16;
    const int nn   = lane & 15;   // B/C/D lane -> N within tile
    const int half = lane >> 4;   // K-half for A/B, M-half for C/D (ISA 7.12.2)

    v8f acc = {};
    const unsigned short* arow = &sA[nn * LDS_STRIDE];
    const unsigned short* bcol = Wp + (size_t)((colBase + nn) * 2 + half) * 16;
    #pragma unroll
    for (int kk = 0; kk < 8; ++kk) {          // K = 256, chunks of 32
        const int kb = kk * 32;
        uint4 alo = *(const uint4*)(arow + kb + half * 16);       // K kb+16*half..+7
        uint4 ahi = *(const uint4*)(arow + kb + half * 16 + 8);   // K kb+16*half+8..+15
        const uint4* bp = (const uint4*)(bcol + (size_t)kk * 8192);
        uint4 blo = bp[0], bhi = bp[1];
        acc = __builtin_amdgcn_wmma_f32_16x16x32_bf16(false, frag_from(alo, ahi),
                                                      false, frag_from(blo, bhi),
                                                      (short)0, acc, false, false);
    }
    // C/D layout: N = lane&15, M = r + 8*(lane>>4).
    float* orow = Hout + (size_t)rowBase * DDIM + colBase + nn;
    if (rowBase + 16 <= n_rows) {             // uniform fast path (always here)
        #pragma unroll
        for (int r = 0; r < 8; ++r) orow[(size_t)(r + 8 * half) * DDIM] = acc[r];
    } else {
        #pragma unroll
        for (int r = 0; r < 8; ++r) {
            int m = r + 8 * half;
            if (rowBase + m < n_rows) orow[(size_t)m * DDIM] = acc[r];
        }
    }
}

// wa_src[d,h] = sum_c W[d, h*64+c] * asrc[h,c]  (same for adst). 2048 outputs.
__global__ void k_project(const float* __restrict__ W, const float* __restrict__ asrc,
                          const float* __restrict__ adst,
                          float* __restrict__ was, float* __restrict__ wad) {
    int tid = blockIdx.x * blockDim.x + threadIdx.x;
    if (tid >= 2048) return;
    const float* a = (tid < 1024) ? asrc : adst;
    float* o       = (tid < 1024) ? was  : wad;
    int t = tid & 1023;
    int d = t >> 2, h = t & 3;
    float s = 0.f;
    #pragma unroll 8
    for (int c = 0; c < 64; ++c) s += W[d * DDIM + h * 64 + c] * a[h * 64 + c];
    o[d * 4 + h] = s;
}

// al[n,h] = x[n,:] . wa[:,h]
__global__ void k_logits(const float* __restrict__ X, const float* __restrict__ wa,
                         float* __restrict__ al, int n) {
    int i = blockIdx.x * blockDim.x + threadIdx.x;
    if (i >= n * 4) return;
    int node = i >> 2, h = i & 3;
    const float* x = X + (size_t)node * DDIM;
    float s = 0.f;
    #pragma unroll 8
    for (int d = 0; d < DDIM; ++d) s += x[d] * wa[d * 4 + h];
    al[i] = s;
}

static __device__ __forceinline__ float lrelu(float v) { return v > 0.f ? v : 0.2f * v; }

__global__ void k_edge_max(const int* __restrict__ eidx, int E,
                           const float* __restrict__ als, const float* __restrict__ ald,
                           unsigned* __restrict__ m) {
    int i = blockIdx.x * blockDim.x + threadIdx.x;
    if (i >= E * 4) return;
    int e = i >> 2, h = i & 3;
    int s = eidx[e], d = eidx[E + e];
    float v = lrelu(als[s * 4 + h] + ald[d * 4 + h]);
    atomicMax(m + d * 4 + h, encf(v));
}

__global__ void k_edge_sum(const int* __restrict__ eidx, int E,
                           const float* __restrict__ als, const float* __restrict__ ald,
                           const unsigned* __restrict__ m, float* __restrict__ den) {
    int i = blockIdx.x * blockDim.x + threadIdx.x;
    if (i >= E * 4) return;
    int e = i >> 2, h = i & 3;
    int s = eidx[e], d = eidx[E + e];
    float v = lrelu(als[s * 4 + h] + ald[d * 4 + h]);
    atomicAdd(den + d * 4 + h, __expf(v - decf(m[d * 4 + h])));
}

// 64 threads/edge; thread l covers 4 channels of head l>>4. Gathers from L2-resident h_src.
__global__ void k_message(const int* __restrict__ eidx, int E,
                          const float* __restrict__ als, const float* __restrict__ ald,
                          const unsigned* __restrict__ mm, const float* __restrict__ den,
                          const float* __restrict__ Hsrc, float* __restrict__ out) {
    int i = blockIdx.x * blockDim.x + threadIdx.x;
    if (i >= E * 64) return;
    int e = i >> 6;
    int l = i & 63;
    int h = l >> 4;
    int ch = l * 4;
    int s = eidx[e], d = eidx[E + e];
    float v = lrelu(als[s * 4 + h] + ald[d * 4 + h]);
    float coeff = __expf(v - decf(mm[d * 4 + h])) / (den[d * 4 + h] + 1e-16f);
    float4 hv = *(const float4*)(Hsrc + (size_t)s * DDIM + ch);
    float* o = out + (size_t)d * DDIM + ch;
    atomicAdd(o + 0, hv.x * coeff);
    atomicAdd(o + 1, hv.y * coeff);
    atomicAdd(o + 2, hv.z * coeff);
    atomicAdd(o + 3, hv.w * coeff);
}

// out = elu(acc + b1[col] (+ b2[col]))
__global__ void k_finalize(float* __restrict__ out, const float* __restrict__ b1,
                           const float* __restrict__ b2, int n) {
    int i = blockIdx.x * blockDim.x + threadIdx.x;
    if (i >= n) return;
    int k = i & 255;
    float v = out[i] + b1[k] + (b2 ? b2[k] : 0.f);
    out[i] = (v > 0.f) ? v : (__expf(v) - 1.f);
}

extern "C" void kernel_launch(void* const* d_in, const int* in_sizes, int n_in,
                              void* d_out, int out_size, void* d_ws, size_t ws_size,
                              hipStream_t stream) {
    const float* x_paper  = (const float*)d_in[0];
    const float* x_author = (const float*)d_in[1];
    const int* e_cites  = (const int*)d_in[2];
    const int* e_writes = (const int*)d_in[3];
    const int* e_wb     = (const int*)d_in[4];
    const int NP = in_sizes[0] / DDIM;
    const int NA = in_sizes[1] / DDIM;
    const int EC = in_sizes[2] / 2;
    const int EW = in_sizes[3] / 2;
    const int EB = in_sizes[4] / 2;

    float* out_p = (float*)d_out;
    float* out_a = out_p + (size_t)NP * DDIM;

    // Workspace, reused sequentially per relation (keeps h_src L2-resident):
    float*          ws   = (float*)d_ws;
    float*          HSRC = ws;                                  // NP*256 f32 (max)
    unsigned short* XB   = (unsigned short*)(HSRC + (size_t)NP * DDIM);  // NP*256 bf16 (max)
    unsigned short* WP   = XB + (size_t)NP * DDIM;              // 65536 bf16
    float*          WAS  = (float*)(WP + 65536);                // 1024
    float*          WAD  = WAS + 1024;                          // 1024
    float*          ALS  = WAD + 1024;                          // NP*4 (max)
    float*          ALD  = ALS + (size_t)NP * 4;                // NP*4 (max)
    unsigned*       MM   = (unsigned*)(ALD + (size_t)NP * 4);   // NP*4 (max)
    float*          DEN  = (float*)(MM + (size_t)NP * 4);       // NP*4 (max)

    int nOut4 = (NP + NA) * (DDIM / 4);
    k_zero<<<(nOut4 + 255) / 256, 256, 0, stream>>>((float*)d_out, nOut4);

    struct Rel { const float* xs; int ns; const float* xd; int nd;
                 const int* e; int E; const float* W; const float* as;
                 const float* ad; float* out; };
    Rel rels[3] = {
        { x_paper,  NP, x_paper,  NP, e_cites,  EC,
          (const float*)d_in[5],  (const float*)d_in[6],  (const float*)d_in[7],  out_p },
        { x_author, NA, x_paper,  NP, e_writes, EW,
          (const float*)d_in[9],  (const float*)d_in[10], (const float*)d_in[11], out_p },
        { x_paper,  NP, x_author, NA, e_wb,     EB,
          (const float*)d_in[13], (const float*)d_in[14], (const float*)d_in[15], out_a },
    };

    for (int r = 0; r < 3; ++r) {
        const Rel& R = rels[r];
        k_init_md<<<(R.nd * 4 + 255) / 256, 256, 0, stream>>>(MM, DEN, R.nd * 4);
        int n2 = R.ns * (DDIM / 2);
        k_cvt_bf16<<<(n2 + 255) / 256, 256, 0, stream>>>(R.xs, (unsigned*)XB, n2);
        k_pack_w<<<16, 256, 0, stream>>>(R.W, WP);
        k_gemm<<<(R.ns + 15) / 16, 512, 0, stream>>>(XB, WP, HSRC, R.ns);
        k_project<<<8, 256, 0, stream>>>(R.W, R.as, R.ad, WAS, WAD);
        k_logits<<<(R.ns * 4 + 255) / 256, 256, 0, stream>>>(R.xs, WAS, ALS, R.ns);
        k_logits<<<(R.nd * 4 + 255) / 256, 256, 0, stream>>>(R.xd, WAD, ALD, R.nd);
        k_edge_max<<<(R.E * 4 + 255) / 256, 256, 0, stream>>>(R.e, R.E, ALS, ALD, MM);
        k_edge_sum<<<(R.E * 4 + 255) / 256, 256, 0, stream>>>(R.e, R.E, ALS, ALD, MM, DEN);
        k_message<<<(R.E * 64 + 255) / 256, 256, 0, stream>>>(R.e, R.E, ALS, ALD, MM, DEN,
                                                              HSRC, R.out);
    }

    k_finalize<<<(NP * DDIM + 255) / 256, 256, 0, stream>>>(
        out_p, (const float*)d_in[8], (const float*)d_in[12], NP * DDIM);
    k_finalize<<<(NA * DDIM + 255) / 256, 256, 0, stream>>>(
        out_a, (const float*)d_in[16], nullptr, NA * DDIM);
}
